// NGCF_30502857736234
// MI455X (gfx1250) — matrix operand order repaired and checked by reading, not compiled, not run on hardware
//
#include <hip/hip_runtime.h>
#include <hip/hip_bf16.h>

// ---------------------------------------------------------------------------
// NGCF forward for MI455X (gfx1250, wave32).
//  - SpMM via f32 global atomics (x + side are L2-resident: 51MB each << 192MB L2)
//  - Dense 128x128 transforms via v_wmma_f32_16x16x32_f16 (f16 in, f32 acc)
//  - Weights staged to LDS once per block via the Tensor Data Mover (TDM),
//    with hardware row padding for bank-conflict-free ds_load_b128 B-fetches
//  - Fused bias + leaky_relu + row-L2-normalize epilogue
// ---------------------------------------------------------------------------

typedef _Float16 v16h __attribute__((ext_vector_type(16)));
typedef _Float16 v8h  __attribute__((ext_vector_type(8)));
typedef _Float16 v4h  __attribute__((ext_vector_type(4)));
typedef float    v8f  __attribute__((ext_vector_type(8)));
typedef unsigned int u32x4 __attribute__((ext_vector_type(4)));
typedef int          i32x8 __attribute__((ext_vector_type(8)));
typedef int          i32x4 __attribute__((ext_vector_type(4)));

#define DIMV 128
#define WROW 136   // padded LDS row for weights: 136 halves = 272B
#define HROW 136   // padded LDS row for activations

#if __has_builtin(__builtin_amdgcn_tensor_load_to_lds) && \
    __has_builtin(__builtin_amdgcn_s_wait_tensorcnt)
#define NGCF_HAS_TDM 1
#else
#define NGCF_HAS_TDM 0
#endif

static __device__ __forceinline__ v16h join8(v8h lo, v8h hi) {
  return __builtin_shufflevector(lo, hi, 0, 1, 2, 3, 4, 5, 6, 7,
                                         8, 9, 10, 11, 12, 13, 14, 15);
}

#if NGCF_HAS_TDM
// Issue one TDM 2D tile load: 128x128 f16, row-major, into LDS at
// lds_byte_off with 16B of padding appended per 256B row (-> 272B LDS rows).
// D# layout per CDNA5 ISA §8.3/8.4.  This toolchain exposes the 6-arg builtin
// (clang-23 form): (u32x4 g0, i32x8 g1, i32x4 g2, i32x4 g3, i32x8 extra, cpol).
static __device__ __forceinline__ void tdm_load_w_tile(const _Float16* gsrc,
                                                       unsigned lds_byte_off) {
  unsigned long long ga = (unsigned long long)(uintptr_t)gsrc;
  u32x4 g0;
  g0[0] = 1u;                                   // count=1, user-mode, no gather
  g0[1] = lds_byte_off;                         // lds_addr (bytes)
  g0[2] = (unsigned)(ga & 0xffffffffu);         // global_addr[31:0]
  g0[3] = (unsigned)((ga >> 32) & 0x01ffffffu)  // global_addr[56:32]
          | 0x80000000u;                        // type=2 ("image") in [127:126]
  i32x8 g1;
  g1[0] = (int)((1u << 16)      // data_size = 1 -> 2 bytes
                | (1u << 20)    // pad_enable
                | (5u << 22)    // pad_interval code 5 -> every 64 DWORDs (256B)
                | (3u << 25));  // pad_amount code 3 -> 4 DWORDs (16B)
  g1[1] = (int)(128u << 16);    // tensor_dim0[15:0] = 128 (bits 79:48)
  g1[2] = (int)(128u << 16);    // tensor_dim0 hi = 0; tensor_dim1 lo = 128
  g1[3] = (int)(128u << 16);    // tensor_dim1 hi = 0; tile_dim0 = 128
  g1[4] = (int)128u;            // tile_dim1 = 128; tile_dim2 = 0 (2D)
  g1[5] = (int)128u;            // tensor_dim0_stride lo = 128 elements
  g1[6] = 0;                    // stride hi / tensor_dim1_stride lo
  g1[7] = 0;
  i32x4 z4 = {0, 0, 0, 0};              // groups 2/3 unused for 2D tiles
  i32x8 z8 = {0, 0, 0, 0, 0, 0, 0, 0};  // extra group (zero)
  __builtin_amdgcn_tensor_load_to_lds(g0, g1, z4, z4, z8, 0);
}
#endif

// ---------------------------------------------------------------------------
// Zero fill (float4 granularity; count is in float4 units)
// ---------------------------------------------------------------------------
__global__ void ngcf_fill_zero(float4* __restrict__ p, long long n4) {
  long long i = (long long)blockIdx.x * blockDim.x + threadIdx.x;
  if (i < n4) p[i] = make_float4(0.f, 0.f, 0.f, 0.f);
}

// ---------------------------------------------------------------------------
// out[:, 0:128] = emb   (one float4 per thread)
// ---------------------------------------------------------------------------
__global__ void ngcf_copy_emb(const float4* __restrict__ emb,
                              float* __restrict__ out, int n, int ostride) {
  long long u = (long long)blockIdx.x * blockDim.x + threadIdx.x;
  long long total = (long long)n * (DIMV / 4);
  if (u >= total) return;
  long long row = u >> 5;  // 32 float4 per row
  int c4 = (int)(u & 31);
  float4 v = emb[row * 32 + c4];
  *(float4*)(out + row * (long long)ostride + c4 * 4) = v;
}

// ---------------------------------------------------------------------------
// Convert both weight tensors f32 -> f16 (layout preserved, row-major [out][in])
// ---------------------------------------------------------------------------
__global__ void ngcf_cvt_weights(const float* __restrict__ w1,
                                 const float* __restrict__ w2,
                                 _Float16* __restrict__ h1,
                                 _Float16* __restrict__ h2, int cnt) {
  int i = blockIdx.x * blockDim.x + threadIdx.x;
  if (i >= cnt) return;
  h1[i] = (_Float16)w1[i];
  h2[i] = (_Float16)w2[i];
}

// ---------------------------------------------------------------------------
// COO SpMM scatter: side[row] += val * x[col].  One wave per edge: 32 lanes x
// float4 cover the 128-wide row; 4 f32 atomics per lane (L2-resident).
// ---------------------------------------------------------------------------
__global__ __launch_bounds__(256) void ngcf_spmm_scatter(
    const int* __restrict__ adj_row, const int* __restrict__ adj_col,
    const float* __restrict__ adj_vals, const float* __restrict__ xin,
    int xstride, float* __restrict__ side, int nnz) {
  int e = blockIdx.x * (blockDim.x >> 5) + (threadIdx.x >> 5);
  int lane = threadIdx.x & 31;
  if (e >= nnz) return;
  int r = adj_row[e];
  int c = adj_col[e];
  float v = adj_vals[e];
  const float4 xv = *(const float4*)(xin + (long long)c * xstride + lane * 4);
  float* d = side + (long long)r * DIMV + lane * 4;
  atomicAdd(d + 0, v * xv.x);
  atomicAdd(d + 1, v * xv.y);
  atomicAdd(d + 2, v * xv.z);
  atomicAdd(d + 3, v * xv.w);
}

// ---------------------------------------------------------------------------
// Fused dense layer.  4 waves / block, 16 rows / wave.
// Dynamic LDS layout (104,448 B):
//   [ W1 : 128 rows x 136 halves ][ W2 : same ][ h1 : 4x16x136 ][ h2 : same ]
// ---------------------------------------------------------------------------
__global__ __launch_bounds__(128) void ngcf_dense(
    const float* __restrict__ xin, int xstride,
    const float* __restrict__ side,
    const _Float16* __restrict__ w1, const float* __restrict__ b1,
    const _Float16* __restrict__ w2, const float* __restrict__ b2,
    float* __restrict__ outp, int ostride, int nrows) {
  extern __shared__ __align__(16) _Float16 smem[];
  _Float16* wlds1 = smem;                       // 128*WROW halves
  _Float16* wlds2 = smem + DIMV * WROW;
  _Float16* h1all = smem + 2 * DIMV * WROW;     // 4*16*HROW halves
  _Float16* h2all = h1all + 4 * 16 * HROW;

  const int wv = threadIdx.x >> 5;
  const int lane = threadIdx.x & 31;
  const int m0 = blockIdx.x * 64 + wv * 16;  // first row of this wave's tile
  const bool active = (m0 < nrows);          // nrows % 16 == 0

  // --- Stage weights into LDS (once per block) ---
#if NGCF_HAS_TDM
  if (wv == 0) {
    tdm_load_w_tile(w1, (unsigned)(uintptr_t)wlds1);
    tdm_load_w_tile(w2, (unsigned)(uintptr_t)wlds2);
  }
#else
  // Cooperative copy producing the identical padded layout.
  for (int u = threadIdx.x; u < (DIMV * DIMV / 8); u += 128) {
    int row = u >> 4;          // 16 v8h units per 128-half row
    int cu = (u & 15) * 8;
    *(v8h*)(wlds1 + row * WROW + cu) = *(const v8h*)(w1 + row * DIMV + cu);
    *(v8h*)(wlds2 + row * WROW + cu) = *(const v8h*)(w2 + row * DIMV + cu);
  }
#endif

  // --- Stage h1 = side + x, h2 = side * x as f16 (per-wave region) ---
  _Float16* h1 = h1all + wv * 16 * HROW;
  _Float16* h2 = h2all + wv * 16 * HROW;
  if (active) {
    const int c0 = lane * 4;
#pragma unroll 4
    for (int i = 0; i < 16; ++i) {
      const long long row = m0 + i;
      const float4 sv = *(const float4*)(side + row * DIMV + c0);
      const float4 xv = *(const float4*)(xin + row * (long long)xstride + c0);
      v4h va = {(_Float16)(sv.x + xv.x), (_Float16)(sv.y + xv.y),
                (_Float16)(sv.z + xv.z), (_Float16)(sv.w + xv.w)};
      v4h vp = {(_Float16)(sv.x * xv.x), (_Float16)(sv.y * xv.y),
                (_Float16)(sv.z * xv.z), (_Float16)(sv.w * xv.w)};
      *(v4h*)(h1 + i * HROW + c0) = va;
      *(v4h*)(h2 + i * HROW + c0) = vp;
    }
  }
#if NGCF_HAS_TDM
  if (wv == 0) __builtin_amdgcn_s_wait_tensorcnt(0);
#endif
  __syncthreads();
  if (!active) return;

  const int hs = lane >> 4;  // half-wave select
  const int mr = lane & 15;  // row within tile (A) / column within tile (B)

  // A fragments (contiguous K order, matching B):
  // lanes 0-15  hold K = kb*32 + [0..15]
  // lanes 16-31 hold K = kb*32 + [16..31]
  v16h A1[4], A2[4];
  {
    const _Float16* r1 = h1 + mr * HROW + hs * 16;
    const _Float16* r2 = h2 + mr * HROW + hs * 16;
#pragma unroll
    for (int kb = 0; kb < 4; ++kb) {
      v8h lo1 = *(const v8h*)(r1 + kb * 32);
      v8h hi1 = *(const v8h*)(r1 + kb * 32 + 8);
      A1[kb] = join8(lo1, hi1);
      v8h lo2 = *(const v8h*)(r2 + kb * 32);
      v8h hi2 = *(const v8h*)(r2 + kb * 32 + 8);
      A2[kb] = join8(lo2, hi2);
    }
  }

  // B fragments in this code's contiguous K order (matching A): lane n (0-15) holds K=kb*32+[0..15]
  // of W row n (contiguous in LDS); lanes 16-31 hold K=kb*32+[16..31].
  v8f acc[8];
#pragma unroll
  for (int n0 = 0; n0 < 8; ++n0) {
    const int col = n0 * 16 + mr;
    const float bsum = b1[col] + b2[col];
    v8f c;
#pragma unroll
    for (int i = 0; i < 8; ++i) c[i] = bsum;

    const _Float16* w1p = wlds1 + col * WROW + hs * 16;
    const _Float16* w2p = wlds2 + col * WROW + hs * 16;
#pragma unroll
    for (int kb = 0; kb < 4; ++kb) {
      const v8h* p1 = (const v8h*)(w1p + kb * 32);
      v16h B1 = join8(p1[0], p1[1]);
      c = __builtin_amdgcn_wmma_f32_16x16x32_f16(false, A1[kb], false, B1,
                                                 (short)0, c, false, false);
      const v8h* p2 = (const v8h*)(w2p + kb * 32);
      v16h B2 = join8(p2[0], p2[1]);
      c = __builtin_amdgcn_wmma_f32_16x16x32_f16(false, A2[kb], false, B2,
                                                 (short)0, c, false, false);
    }
    acc[n0] = c;
  }

  // Epilogue: leaky_relu then per-row sum of squares.
  // C/D layout: VGPR r, lane l -> element (M = r + 8*(l>=16), N = l%16).
  float ss[8];
#pragma unroll
  for (int r = 0; r < 8; ++r) ss[r] = 0.f;
#pragma unroll
  for (int n0 = 0; n0 < 8; ++n0) {
#pragma unroll
    for (int r = 0; r < 8; ++r) {
      float y = acc[n0][r];
      y = (y > 0.f) ? y : 0.01f * y;
      acc[n0][r] = y;
      ss[r] += y * y;
    }
  }
  // Reduce across the 16 lanes of each half-wave (masks 1,2,4,8 stay in-half).
#pragma unroll
  for (int r = 0; r < 8; ++r) {
    float s = ss[r];
    s += __shfl_xor(s, 1);
    s += __shfl_xor(s, 2);
    s += __shfl_xor(s, 4);
    s += __shfl_xor(s, 8);
    ss[r] = 1.0f / fmaxf(sqrtf(s), 1e-12f);
  }
  // Scale + store (16 contiguous floats per half-wave per tile-row).
#pragma unroll
  for (int n0 = 0; n0 < 8; ++n0) {
#pragma unroll
    for (int r = 0; r < 8; ++r) {
      const long long row = m0 + r + 8 * hs;
      outp[row * (long long)ostride + n0 * 16 + mr] = acc[n0][r] * ss[r];
    }
  }
}

// ---------------------------------------------------------------------------
// Host-side orchestration (graph-capture safe: only kernel launches on stream)
// ---------------------------------------------------------------------------
extern "C" void kernel_launch(void* const* d_in, const int* in_sizes, int n_in,
                              void* d_out, int out_size, void* d_ws,
                              size_t ws_size, hipStream_t stream) {
  const int* adj_row = (const int*)d_in[0];
  const int* adj_col = (const int*)d_in[1];
  const float* adj_vals = (const float*)d_in[2];
  const float* emb = (const float*)d_in[3];
  const float* W1_w = (const float*)d_in[4];
  const float* W1_b = (const float*)d_in[5];
  const float* W2_w = (const float*)d_in[6];
  const float* W2_b = (const float*)d_in[7];

  const int nnz = in_sizes[0];
  const int n = in_sizes[3] / DIMV;       // number of nodes
  const int layers = in_sizes[5] / DIMV;  // LAYERS
  const int ostride = DIMV * (layers + 1);

  // Workspace layout: [ side : N*128 f32 ][ W1 f16 ][ W2 f16 ]
  float* side = (float*)d_ws;
  size_t side_bytes = (size_t)n * DIMV * sizeof(float);
  size_t woff = (side_bytes + 255) & ~(size_t)255;
  _Float16* wh1 = (_Float16*)((char*)d_ws + woff);
  _Float16* wh2 = wh1 + (size_t)layers * DIMV * DIMV;
  float* out = (float*)d_out;

  // Dynamic LDS: padded W1 + W2 + per-wave h1/h2 staging.
  const size_t smem_bytes =
      (size_t)(2 * DIMV * WROW + 2 * 4 * 16 * HROW) * sizeof(_Float16);

  // 1) Weights -> f16 (tiny, stays in L2)
  {
    int cnt = layers * DIMV * DIMV;
    ngcf_cvt_weights<<<(cnt + 255) / 256, 256, 0, stream>>>(W1_w, W2_w, wh1,
                                                            wh2, cnt);
  }
  // 2) out[:, 0:128] = emb
  {
    long long units = (long long)n * (DIMV / 4);
    ngcf_copy_emb<<<(int)((units + 255) / 256), 256, 0, stream>>>(
        (const float4*)emb, out, n, ostride);
  }
  // 3) Layers
  for (int k = 0; k < layers; ++k) {
    const float* xin = (k == 0) ? emb : (out + (size_t)k * DIMV);
    int xstride = (k == 0) ? DIMV : ostride;

    long long n4 = (long long)n * (DIMV / 4);
    ngcf_fill_zero<<<(int)((n4 + 255) / 256), 256, 0, stream>>>((float4*)side,
                                                                n4);
    ngcf_spmm_scatter<<<(nnz + 7) / 8, 256, 0, stream>>>(
        adj_row, adj_col, adj_vals, xin, xstride, side, nnz);
    ngcf_dense<<<(n + 63) / 64, 128, smem_bytes, stream>>>(
        xin, xstride, side, wh1 + (size_t)k * DIMV * DIMV,
        W1_b + (size_t)k * DIMV, wh2 + (size_t)k * DIMV * DIMV,
        W2_b + (size_t)k * DIMV, out + (size_t)(k + 1) * DIMV, ostride, n);
  }
}
